// SequentialReader_62130996903986
// MI455X (gfx1250) — matrix operand order repaired and checked by reading, not compiled, not run on hardware
//
#include <hip/hip_runtime.h>
#include <hip/hip_bf16.h>
#include <stdint.h>

// Problem constants (from reference): B=1024, T=64, D=320, H=512
#define B_N   1024
#define T_N   64
#define D_N   320
#define H_N   512
#define G4_N  2048          // 4*H
#define BM    16            // batch rows per block
#define NTHREADS 256        // 8 wave32s
#define KCX   (D_N / 32)    // 10 k-chunks over D
#define KCH   (H_N / 32)    // 16 k-chunks over H

// LDS layout (dynamic shared mem, padded strides to dodge bank conflicts)
#define XS    328           // sX row stride (ushorts), 320 + 8 pad
#define HS    520           // sH row stride (ushorts), 512 + 8 pad
#define GS    2052          // gBuf row stride (floats), 2048 + 4 pad
#define SX_BYTES (BM * XS * 2)          // 10496
#define SH_BYTES (BM * HS * 2)          // 16640
#define G_BYTES  (BM * GS * 4)          // 131328
#define SMEM_BYTES (SX_BYTES + SH_BYTES + G_BYTES)   // 158464

typedef __attribute__((ext_vector_type(16))) __bf16 v16bf;
typedef __attribute__((ext_vector_type(8)))  float  v8f;

union Frag {
  v16bf v;
  uint4 q[2];
};

__device__ __forceinline__ unsigned short f2bf(float f) {
  unsigned int u = __float_as_uint(f);
  unsigned int r = (u + 0x7FFFu + ((u >> 16) & 1u)) >> 16;  // RNE
  return (unsigned short)r;
}

__device__ __forceinline__ float sigf(float x) {
  return 1.0f / (1.0f + __expf(-x));
}

// A fragment (16x32 bf16) from LDS, per CDNA5 16-bit A layout:
// lanes 0-15 : M=lane,    K = k0+0..7  and k0+16..23
// lanes16-31 : M=lane-16, K = k0+8..15 and k0+24..31
__device__ __forceinline__ v16bf load_a_lds(const unsigned short* sA, int stride,
                                            int k0, int lane) {
  int mm = lane & 15;
  int kb = (lane >> 4) << 3;
  const unsigned short* p = sA + mm * stride + k0 + kb;
  Frag f;
  f.q[0] = *(const uint4*)(p);        // 8 bf16
  f.q[1] = *(const uint4*)(p + 16);   // 8 bf16 at +16 elems (+32B)
  return f.v;
}

// B fragment (32x16 bf16) from pre-packed global weights: per lane 16
// contiguous bf16 (32B) -> two global_load_b128, fully coalesced per wave.
__device__ __forceinline__ v16bf load_b_glb(const unsigned short* __restrict__ pack,
                                            int fragIdx, int lane) {
  const unsigned short* p = pack + ((size_t)fragIdx << 9) + (lane << 4);
  Frag f;
  f.q[0] = *(const uint4*)(p);
  f.q[1] = *(const uint4*)(p + 8);
  return f.v;
}

__device__ __forceinline__ v8f wmma_bf16(v16bf a, v16bf b, v8f c) {
  return __builtin_amdgcn_wmma_f32_16x16x32_bf16(false, a, false, b,
                                                 (short)0, c, false, false);
}

// ---------------------------------------------------------------------------
// Pack W[N][K] fp32 row-major into bf16 WMMA B-fragments.
// frag f = ntile*(K/32)+kchunk ; elem (lane,i):
//   n = ntile*16 + (lane&15), k = kchunk*32 + (lane>>4)*8 + (i&7) + (i>>3)*16
// ---------------------------------------------------------------------------
__global__ void pack_weight_kernel(const float* __restrict__ W,
                                   unsigned short* __restrict__ out,
                                   int N, int K) {
  int total = N * K;
  int KC = K >> 5;
  for (int idx = blockIdx.x * blockDim.x + threadIdx.x; idx < total;
       idx += gridDim.x * blockDim.x) {
    int i    = idx & 15;
    int lane = (idx >> 4) & 31;
    int f    = idx >> 9;
    int ntile = f / KC;
    int kch   = f - ntile * KC;
    int n = (ntile << 4) + (lane & 15);
    int k = (kch << 5) + ((lane >> 4) << 3) + (i & 7) + ((i >> 3) << 4);
    out[idx] = f2bf(W[(size_t)n * K + k]);
  }
}

__global__ void combine_bias_kernel(const float* __restrict__ b_ih,
                                    const float* __restrict__ b_hh,
                                    const float* __restrict__ rw_b,
                                    const float* __restrict__ hw_b,
                                    float* __restrict__ bsum,
                                    float* __restrict__ rbsum) {
  int i = blockIdx.x * blockDim.x + threadIdx.x;
  if (i < G4_N) bsum[i] = b_ih[i] + b_hh[i];
  if (i < H_N)  rbsum[i] = rw_b[i] + hw_b[i];
}

// 4-tile GEMM strip with explicit double buffering: fragments for chunk kc+1
// (A from LDS, 4 B's from global) are live while the 4 WMMAs of chunk kc
// issue, so waits become partial (loadcnt<=N) instead of full drains.
__device__ __forceinline__ void gemm_strip4(
    const unsigned short* sX, const unsigned short* sH,
    const unsigned short* __restrict__ xw_pack,   // [Ntiles][KCX] frags
    const unsigned short* __restrict__ hw_pack,   // [Ntiles][KCH] frags
    const float* __restrict__ bias,               // per-column bias
    float* gBuf, int nt0, int lane, int mb, int nc)
{
  v8f acc[4];
#pragma unroll
  for (int u = 0; u < 4; ++u) {
    float bsv = bias[((nt0 + u) << 4) + nc];
    acc[u] = (v8f){bsv, bsv, bsv, bsv, bsv, bsv, bsv, bsv};
  }

  // Prologue: fragments for X-phase chunk 0
  v16bf a0 = load_a_lds(sX, XS, 0, lane);
  v16bf b0[4];
#pragma unroll
  for (int u = 0; u < 4; ++u)
    b0[u] = load_b_glb(xw_pack, (nt0 + u) * KCX, lane);

  // X phase (D contraction), prefetching next chunk (or first H chunk)
#pragma unroll 2
  for (int kc = 0; kc < KCX; ++kc) {
    v16bf a1;
    v16bf b1[4];
    if (kc + 1 < KCX) {
      a1 = load_a_lds(sX, XS, (kc + 1) << 5, lane);
#pragma unroll
      for (int u = 0; u < 4; ++u)
        b1[u] = load_b_glb(xw_pack, (nt0 + u) * KCX + kc + 1, lane);
    } else {
      a1 = load_a_lds(sH, HS, 0, lane);
#pragma unroll
      for (int u = 0; u < 4; ++u)
        b1[u] = load_b_glb(hw_pack, (nt0 + u) * KCH, lane);
    }
#pragma unroll
    for (int u = 0; u < 4; ++u)
      acc[u] = wmma_bf16(a0, b0[u], acc[u]);
    a0 = a1;
#pragma unroll
    for (int u = 0; u < 4; ++u) b0[u] = b1[u];
  }

  // H phase (H contraction)
#pragma unroll 2
  for (int kc = 0; kc < KCH; ++kc) {
    v16bf a1;
    v16bf b1[4];
    if (kc + 1 < KCH) {
      a1 = load_a_lds(sH, HS, (kc + 1) << 5, lane);
#pragma unroll
      for (int u = 0; u < 4; ++u)
        b1[u] = load_b_glb(hw_pack, (nt0 + u) * KCH + kc + 1, lane);
    }
#pragma unroll
    for (int u = 0; u < 4; ++u)
      acc[u] = wmma_bf16(a0, b0[u], acc[u]);
    if (kc + 1 < KCH) {
      a0 = a1;
#pragma unroll
      for (int u = 0; u < 4; ++u) b0[u] = b1[u];
    }
  }

#pragma unroll
  for (int u = 0; u < 4; ++u) {
    int n0 = (nt0 + u) << 4;
#pragma unroll
    for (int q = 0; q < 8; ++q)
      gBuf[(mb + q) * GS + n0 + nc] = acc[u][q];
  }
}

// ---------------------------------------------------------------------------
// Fused persistent LSTM: 64 blocks x 16 batch rows, all 64 timesteps,
// h/c state in registers, bf16 WMMA GEMMs, fc dot accumulated on the fly.
// ---------------------------------------------------------------------------
__global__ __launch_bounds__(NTHREADS)
void lstm_fused_kernel(const float* __restrict__ x,        // [B,T,D]
                       const float* __restrict__ h0,       // [B,H]
                       const float* __restrict__ c0,       // [B,H]
                       const unsigned short* __restrict__ wih_p,
                       const unsigned short* __restrict__ whh_p,
                       const unsigned short* __restrict__ rw_p,
                       const unsigned short* __restrict__ hw_p,
                       const float* __restrict__ bsum,     // [4H]
                       const float* __restrict__ rbsum,    // [H]
                       const float* __restrict__ fcW,      // [T*H]
                       const float* __restrict__ fcb,      // [1]
                       float* __restrict__ out)            // [B]
{
  extern __shared__ char smem[];
  unsigned short* sX  = (unsigned short*)smem;                       // [BM][XS] bf16
  unsigned short* sH  = (unsigned short*)(smem + SX_BYTES);          // [BM][HS] bf16
  float*          gBuf = (float*)(smem + SX_BYTES + SH_BYTES);       // [BM][GS] f32

  const int tid  = threadIdx.x;
  const int lane = tid & 31;
  const int wv   = tid >> 5;          // wave id 0..7
  const int b0   = blockIdx.x * BM;

  // Elementwise ownership: row m = tid/16, column set {r + 16*s : s=0..31}
  const int m = tid >> 4;
  const int r = tid & 15;

  float hreg[32], creg[32];
#pragma unroll 4
  for (int s = 0; s < 32; ++s) {
    int j = r + (s << 4);
    float hv = h0[(size_t)(b0 + m) * H_N + j];
    float cv = c0[(size_t)(b0 + m) * H_N + j];
    hreg[s] = hv;
    creg[s] = cv;
    sH[m * HS + j] = f2bf(hv);
  }
  float facc = 0.0f;
  __syncthreads();

  const int mb = (lane >> 4) << 3;    // C-tile row base per lane half
  const int nc = lane & 15;           // C-tile col within tile

  for (int t = 0; t < T_N; ++t) {
    // ---- stage x_t tile (fp32 -> bf16 LDS), coalesced ----
    for (int idx = tid; idx < BM * D_N; idx += NTHREADS) {
      int mm = idx / D_N;
      int d  = idx - mm * D_N;
      sX[mm * XS + d] =
          f2bf(x[((size_t)(b0 + mm) * T_N + t) * D_N + d]);
    }
    __syncthreads();

    // ---- GEMM1: gates[16][2048] = X@W_ih^T + H@W_hh^T + (b_ih+b_hh) ----
    // 128 n-tiles -> 8 waves x 4 strips of 4 tiles
    for (int g = 0; g < 4; ++g) {
      int nt0 = wv * 16 + g * 4;
      gemm_strip4(sX, sH, wih_p, whh_p, bsum, gBuf, nt0, lane, mb, nc);
    }
    __syncthreads();

    // ---- elementwise LSTM cell (c,h in registers) ----
#pragma unroll 4
    for (int s = 0; s < 32; ++s) {
      int j = r + (s << 4);
      float gi = gBuf[m * GS + j];
      float gf = gBuf[m * GS + H_N + j];
      float gg = gBuf[m * GS + 2 * H_N + j];
      float go = gBuf[m * GS + 3 * H_N + j];
      float cn = sigf(gf) * creg[s] + sigf(gi) * tanhf(gg);
      float hn = sigf(go) * tanhf(cn);
      creg[s] = cn;
      hreg[s] = hn;
      sH[m * HS + j] = f2bf(hn);
    }
    __syncthreads();

    // ---- GEMM2: r[16][512] = X@rw_W^T + Hnew@hw_W^T + (rw_b+hw_b) ----
    // 32 n-tiles -> 8 waves x 1 strip of 4 tiles
    gemm_strip4(sX, sH, rw_p, hw_p, rbsum, gBuf, wv * 4, lane, mb, nc);
    __syncthreads();

    // ---- gated output folded straight into the fc dot product ----
#pragma unroll 4
    for (int s = 0; s < 32; ++s) {
      int j = r + (s << 4);
      float gate = sigf(gBuf[m * GS + j]);
      facc += hreg[s] * gate * fcW[t * H_N + j];
    }
    __syncthreads();   // protect sX/gBuf reuse next iteration
  }

  // ---- reduce 16 partial dots per batch row ----
  float* red = gBuf;
  red[tid] = facc;
  __syncthreads();
  if (tid < BM) {
    float ssum = 0.0f;
#pragma unroll
    for (int i = 0; i < 16; ++i) ssum += red[tid * 16 + i];
    out[b0 + tid] = ssum + fcb[0];
  }
}

// ---------------------------------------------------------------------------
extern "C" void kernel_launch(void* const* d_in, const int* in_sizes, int n_in,
                              void* d_out, int out_size, void* d_ws, size_t ws_size,
                              hipStream_t stream) {
  (void)in_sizes; (void)n_in; (void)out_size; (void)ws_size;

  const float* r_emb = (const float*)d_in[0];
  const float* h0    = (const float*)d_in[1];
  const float* c0    = (const float*)d_in[2];
  const float* W_ih  = (const float*)d_in[3];
  const float* W_hh  = (const float*)d_in[4];
  const float* b_ih  = (const float*)d_in[5];
  const float* b_hh  = (const float*)d_in[6];
  const float* rw_W  = (const float*)d_in[7];
  const float* rw_b  = (const float*)d_in[8];
  const float* hw_W  = (const float*)d_in[9];
  const float* hw_b  = (const float*)d_in[10];
  const float* fc_W  = (const float*)d_in[11];
  const float* fc_b  = (const float*)d_in[12];

  // Workspace layout (bf16 packed weights + combined biases), ~4.3 MB
  char* ws = (char*)d_ws;
  size_t off = 0;
  unsigned short* wih_p = (unsigned short*)(ws + off); off += (size_t)G4_N * D_N * 2; // 1.31MB
  unsigned short* whh_p = (unsigned short*)(ws + off); off += (size_t)G4_N * H_N * 2; // 2.10MB
  unsigned short* rw_p  = (unsigned short*)(ws + off); off += (size_t)H_N  * D_N * 2; // 0.33MB
  unsigned short* hw_p  = (unsigned short*)(ws + off); off += (size_t)H_N  * H_N * 2; // 0.52MB
  float* bsum  = (float*)(ws + off); off += (size_t)G4_N * 4;
  float* rbsum = (float*)(ws + off); off += (size_t)H_N * 4;

  // Pack weights into WMMA B-fragment layout (bf16)
  {
    int tot, grid;
    tot = G4_N * D_N; grid = (tot + 255) / 256;
    pack_weight_kernel<<<grid, 256, 0, stream>>>(W_ih, wih_p, G4_N, D_N);
    tot = G4_N * H_N; grid = (tot + 255) / 256;
    pack_weight_kernel<<<grid, 256, 0, stream>>>(W_hh, whh_p, G4_N, H_N);
    tot = H_N * D_N;  grid = (tot + 255) / 256;
    pack_weight_kernel<<<grid, 256, 0, stream>>>(rw_W, rw_p, H_N, D_N);
    tot = H_N * H_N;  grid = (tot + 255) / 256;
    pack_weight_kernel<<<grid, 256, 0, stream>>>(hw_W, hw_p, H_N, H_N);
  }
  combine_bias_kernel<<<(G4_N + 255) / 256, 256, 0, stream>>>(
      b_ih, b_hh, rw_b, hw_b, bsum, rbsum);

  // Fused persistent recurrent kernel: one block per 16 batch rows
  lstm_fused_kernel<<<B_N / BM, NTHREADS, SMEM_BYTES, stream>>>(
      r_emb, h0, c0, wih_p, whh_p, rw_p, hw_p, bsum, rbsum,
      fc_W, fc_b, (float*)d_out);
}
